// Attention_22239340658869
// MI455X (gfx1250) — compile-verified, matrix operand order checked by reading, not run on hardware
//
#include <hip/hip_runtime.h>
#include <hip/hip_bf16.h>

#define S_LEN 2048
#define H_DIM 2048
#define NHEAD 32
#define KVH   8
#define GRP   4
#define HD    64
#define QKV_N 3072

typedef __attribute__((ext_vector_type(16))) __bf16 bf16x16;
typedef __attribute__((ext_vector_type(8)))  float  f32x8;
typedef __attribute__((ext_vector_type(4)))  unsigned int u32x4;
typedef __attribute__((ext_vector_type(8)))  int          i32x8;
typedef __attribute__((ext_vector_type(4)))  int          i32x4;

#if __has_builtin(__builtin_amdgcn_tensor_load_to_lds) && \
    __has_builtin(__builtin_amdgcn_s_wait_tensorcnt)
#define USE_TDM 1
#else
#define USE_TDM 0
#endif

__device__ __forceinline__ unsigned short f2bf(float f) {
  unsigned int u = __float_as_uint(f);
  unsigned int r = u + 0x7FFFu + ((u >> 16) & 1u);
  return (unsigned short)(r >> 16);
}

__device__ __forceinline__ f32x8 zero8() {
  f32x8 z;
#pragma unroll
  for (int i = 0; i < 8; ++i) z[i] = 0.f;
  return z;
}

__device__ __forceinline__ f32x8 wmma_bf16(bf16x16 a, bf16x16 b, f32x8 c) {
  return __builtin_amdgcn_wmma_f32_16x16x32_bf16(false, a, false, b, (short)0, c,
                                                 false, false);
}

// Gather a 16x32 bf16 WMMA fragment (A-style, row-major source) from LDS.
// Lane l: row = row_base + (l&15); lane-half selects K sub-blocks per ISA layout.
__device__ __forceinline__ bf16x16 load_frag(const unsigned short* lds, int row_base,
                                             int k_base, int stride) {
  const int lane = threadIdx.x & 31;
  const int hlf = lane >> 4;
  const unsigned short* p = lds + (row_base + (lane & 15)) * stride + k_base + hlf * 8;
  union { bf16x16 v; unsigned int u[8]; } fr;
#pragma unroll
  for (int i = 0; i < 4; ++i) {
    fr.u[i]     = *(const unsigned int*)(p + 2 * i);
    fr.u[4 + i] = *(const unsigned int*)(p + 16 + 2 * i);
  }
  return fr.v;
}

#if USE_TDM
__device__ __forceinline__ unsigned lds_off(const void* p) {
  return (unsigned)(unsigned long long)p;
}

// TDM: DMA a tile_d1 x tile_d0 (rows x row-elems) bf16 tile into LDS.
// LDS gets 1 dword of pad per 32 dwords (64 bf16) -> padded row stride 66.
__device__ __forceinline__ void tdm_load_2d(const unsigned short* gptr, unsigned ldsb,
                                            int tensor_d0, int tensor_d1,
                                            int tile_d0, int tile_d1, int stride0) {
  unsigned long long ga = (unsigned long long)gptr;
  u32x4 g0;
  g0[0] = 1u;                                   // count=1, user descriptor
  g0[1] = ldsb;                                 // lds byte address
  g0[2] = (unsigned)(ga & 0xFFFFFFFFu);         // global addr [31:0]
  g0[3] = (unsigned)((ga >> 32) & 0x1FFFFFFu) | (2u << 30);  // [56:32] | type=2
  i32x8 g1;
  g1[0] = (1 << 16) | (1 << 20) | (4 << 22);    // data_size=2B | pad_en | every 32 dw
  g1[1] = (tensor_d0 & 0xFFFF) << 16;           // atomic_barrier=0 | dim0[15:0]
  g1[2] = (int)(((unsigned)tensor_d0 >> 16) | ((unsigned)(tensor_d1 & 0xFFFF) << 16));
  g1[3] = (int)(((unsigned)tensor_d1 >> 16) | ((unsigned)tile_d0 << 16));
  g1[4] = tile_d1;                              // tile_dim1 | tile_dim2=0
  g1[5] = stride0;                              // dim0 stride low 32 (data_size units)
  g1[6] = 0;
  g1[7] = 0;
  i32x4 z4;
  z4[0] = 0; z4[1] = 0; z4[2] = 0; z4[3] = 0;
  i32x8 z8;
#pragma unroll
  for (int i = 0; i < 8; ++i) z8[i] = 0;
  __builtin_amdgcn_tensor_load_to_lds(g0, g1, z4, z4, z8, 0);
}
#endif

__global__ void cvt_f32_bf16_kernel(const float* __restrict__ src,
                                    unsigned short* __restrict__ dst, int n) {
  int i = blockIdx.x * blockDim.x + threadIdx.x;
  if (i < n) dst[i] = f2bf(src[i]);
}

#define LDP 66  // padded LDS row stride (64 data + 2 pad bf16 = 33 dwords)

__device__ __forceinline__ void gemm_compute(const unsigned short* Ab,
                                             const unsigned short* Wb, f32x8 acc[4],
                                             int wave_m, int wave_n) {
#pragma unroll
  for (int ks = 0; ks < 2; ++ks) {
    bf16x16 a = load_frag(Ab, wave_m * 16, ks * 32, LDP);
#pragma unroll
    for (int t = 0; t < 4; ++t) {
      bf16x16 b = load_frag(Wb, (wave_n * 4 + t) * 16, ks * 32, LDP);
      acc[t] = wmma_bf16(a, b, acc[t]);
    }
  }
}

// C[M,N] = A[M,K] * W[N,K]^T, bf16 in, f32 out. 64x128 block tile, K-step 64,
// TDM double-buffered LDS staging.
__global__ __launch_bounds__(256) void gemm_bf16_kernel(
    const unsigned short* __restrict__ A, const unsigned short* __restrict__ W,
    float* __restrict__ C, int M, int N, int K) {
  __shared__ unsigned short As[2][64 * LDP];
  __shared__ unsigned short Ws[2][128 * LDP];
  const int tid = threadIdx.x;
  const int lane = tid & 31, wv = tid >> 5;
  const int hlf = lane >> 4, nl = lane & 15;
  const int wave_m = wv & 3;
  const int wave_n = wv >> 2;
  const int m0 = blockIdx.y * 64;
  const int n0 = blockIdx.x * 128;

  f32x8 acc[4];
#pragma unroll
  for (int t = 0; t < 4; ++t) acc[t] = zero8();

  const int nk = K >> 6;
#if USE_TDM
  if (wv == 0) tdm_load_2d(A + (size_t)m0 * K, lds_off(As[0]), K, M, 64, 64, K);
  if (wv == 1) tdm_load_2d(W + (size_t)n0 * K, lds_off(Ws[0]), K, N, 64, 128, K);
  for (int i = 0; i < nk; ++i) {
    if (wv < 2) __builtin_amdgcn_s_wait_tensorcnt(0);
    __syncthreads();  // tile i visible; all waves done reading buffer (i&1)^1
    if (i + 1 < nk) {
      if (wv == 0)
        tdm_load_2d(A + (size_t)m0 * K + (i + 1) * 64, lds_off(As[(i + 1) & 1]), K, M,
                    64, 64, K);
      if (wv == 1)
        tdm_load_2d(W + (size_t)n0 * K + (i + 1) * 64, lds_off(Ws[(i + 1) & 1]), K, N,
                    64, 128, K);
    }
    gemm_compute(As[i & 1], Ws[i & 1], acc, wave_m, wave_n);
  }
#else
  for (int i = 0; i < nk; ++i) {
    const int k0 = i * 64;
    {  // A tile 64x64: 16 bf16 per thread
      int idx = tid * 16;
      int row = idx >> 6, col = idx & 63;
      const uint4 v0 = *(const uint4*)(A + (size_t)(m0 + row) * K + k0 + col);
      const uint4 v1 = *(const uint4*)(A + (size_t)(m0 + row) * K + k0 + col + 8);
      unsigned int* d = (unsigned int*)(As[0] + row * LDP + col);
      d[0] = v0.x; d[1] = v0.y; d[2] = v0.z; d[3] = v0.w;
      d[4] = v1.x; d[5] = v1.y; d[6] = v1.z; d[7] = v1.w;
    }
#pragma unroll
    for (int rep = 0; rep < 2; ++rep) {  // W tile 128x64: 32 bf16 per thread
      int idx = tid * 16 + rep * 4096;
      int row = idx >> 6, col = idx & 63;
      const uint4 v0 = *(const uint4*)(W + (size_t)(n0 + row) * K + k0 + col);
      const uint4 v1 = *(const uint4*)(W + (size_t)(n0 + row) * K + k0 + col + 8);
      unsigned int* d = (unsigned int*)(Ws[0] + row * LDP + col);
      d[0] = v0.x; d[1] = v0.y; d[2] = v0.z; d[3] = v0.w;
      d[4] = v1.x; d[5] = v1.y; d[6] = v1.z; d[7] = v1.w;
    }
    __syncthreads();
    gemm_compute(As[0], Ws[0], acc, wave_m, wave_n);
    __syncthreads();
  }
#endif

#pragma unroll
  for (int t = 0; t < 4; ++t) {
    int n = n0 + (wave_n * 4 + t) * 16 + nl;
#pragma unroll
    for (int r = 0; r < 8; ++r) {
      int m = m0 + wave_m * 16 + r + hlf * 8;
      C[(size_t)m * N + n] = acc[t][r];
    }
  }
}

// Split qkv, apply RoPE to q/k, convert to bf16. q: [S][NH*HD]; k,v: [KVH][S][HD]
__global__ void rope_split_kernel(const float* __restrict__ qkv,
                                  unsigned short* __restrict__ qb,
                                  unsigned short* __restrict__ kb,
                                  unsigned short* __restrict__ vb) {
  int i = blockIdx.x * blockDim.x + threadIdx.x;
  int j = i & 31;
  int h = (i >> 5) % 48;  // 0..31 q heads, 32..39 k heads, 40..47 v heads
  int s = i / (48 * 32);
  if (s >= S_LEN) return;
  const float* base = qkv + (size_t)s * QKV_N;
  float inv = __powf(10000.f, -(float)j * (1.f / 32.f));
  float ang = (float)s * inv;
  float sn, cs;
  __sincosf(ang, &sn, &cs);
  if (h < NHEAD) {
    float x1 = base[h * HD + j], x2 = base[h * HD + j + 32];
    unsigned short* o = qb + (size_t)s * (NHEAD * HD) + h * HD;
    o[j] = f2bf(x1 * cs - x2 * sn);
    o[j + 32] = f2bf(x2 * cs + x1 * sn);
  } else if (h < NHEAD + KVH) {
    int kh = h - NHEAD;
    float x1 = base[NHEAD * HD + kh * HD + j];
    float x2 = base[NHEAD * HD + kh * HD + j + 32];
    unsigned short* o = kb + ((size_t)kh * S_LEN + s) * HD;
    o[j] = f2bf(x1 * cs - x2 * sn);
    o[j + 32] = f2bf(x2 * cs + x1 * sn);
  } else {
    int kh = h - NHEAD - KVH;
    float x1 = base[(NHEAD + KVH) * HD + kh * HD + j];
    float x2 = base[(NHEAD + KVH) * HD + kh * HD + j + 32];
    unsigned short* o = vb + ((size_t)kh * S_LEN + s) * HD;
    o[j] = f2bf(x1);
    o[j + 32] = f2bf(x2);
  }
}

// Causal flash attention: 64 queries x 1 head per block; online softmax in LDS;
// QK^T and PV via v_wmma_f32_16x16x32_bf16; Q/K tiles staged via TDM.
__global__ __launch_bounds__(256) void flash_attn_kernel(
    const unsigned short* __restrict__ qb, const unsigned short* __restrict__ kb,
    const unsigned short* __restrict__ vb, unsigned short* __restrict__ ob) {
  __shared__ unsigned short Qs[64 * LDP];
  __shared__ unsigned short Ks[64 * LDP];
  __shared__ unsigned short Vts[64 * LDP];  // transposed: [d][key]
  __shared__ unsigned short Ps[64 * LDP];
  __shared__ float Sc[64 * 65];
  __shared__ float row_max[64], row_sum[64], row_scale[64];

  const int tid = threadIdx.x;
  const int lane = tid & 31, wv = tid >> 5;
  const int hlf = lane >> 4, nl = lane & 15;
  const int wave_m = wv & 3, wave_n = wv >> 2;
  const int h = blockIdx.y;
  const int kh = h >> 2;  // GQA 4:1
  const int q0 = blockIdx.x * 64;

#if USE_TDM
  if (wv == 0)
    tdm_load_2d(qb + (size_t)q0 * (NHEAD * HD) + h * HD, lds_off(Qs), NHEAD * HD,
                S_LEN, 64, 64, NHEAD * HD);
#else
  {  // load Q block (64x64 bf16), 16 elems per thread
    int idx = tid * 16;
    int row = idx >> 6, col = idx & 63;
    const unsigned short* src = qb + (size_t)(q0 + row) * (NHEAD * HD) + h * HD + col;
    const uint4 v0 = *(const uint4*)(src);
    const uint4 v1 = *(const uint4*)(src + 8);
    unsigned int* d = (unsigned int*)(Qs + row * LDP + col);
    d[0] = v0.x; d[1] = v0.y; d[2] = v0.z; d[3] = v0.w;
    d[4] = v1.x; d[5] = v1.y; d[6] = v1.z; d[7] = v1.w;
  }
#endif
  if (tid < 64) {
    row_max[tid] = -__builtin_inff();
    row_sum[tid] = 0.f;
  }
#if USE_TDM
  if (wv == 0) __builtin_amdgcn_s_wait_tensorcnt(0);
#endif
  __syncthreads();

  const bf16x16 aq0 = load_frag(Qs, wave_m * 16, 0, LDP);
  const bf16x16 aq1 = load_frag(Qs, wave_m * 16, 32, LDP);
  f32x8 o0 = zero8(), o1 = zero8();

  const int nkt = blockIdx.x + 1;  // causal: key tiles 0..query tile
  for (int kt = 0; kt < nkt; ++kt) {
    const int key0 = kt * 64;
#if USE_TDM
    if (wv == 0)
      tdm_load_2d(kb + ((size_t)kh * S_LEN + key0) * HD, lds_off(Ks), HD, S_LEN, 64,
                  64, HD);
#endif
    {  // V tile transposed (manual; TDM cannot transpose), and K tile on fallback
      int idx = tid * 16;
      int row = idx >> 6, col = idx & 63;
#if !USE_TDM
      const unsigned short* ksrc = kb + ((size_t)kh * S_LEN + key0 + row) * HD + col;
      const uint4 v0 = *(const uint4*)(ksrc);
      const uint4 v1 = *(const uint4*)(ksrc + 8);
      unsigned int* d = (unsigned int*)(Ks + row * LDP + col);
      d[0] = v0.x; d[1] = v0.y; d[2] = v0.z; d[3] = v0.w;
      d[4] = v1.x; d[5] = v1.y; d[6] = v1.z; d[7] = v1.w;
#endif
      const unsigned short* vsrc = vb + ((size_t)kh * S_LEN + key0 + row) * HD + col;
#pragma unroll
      for (int e = 0; e < 16; ++e) Vts[(col + e) * LDP + row] = vsrc[e];
    }
#if USE_TDM
    if (wv == 0) __builtin_amdgcn_s_wait_tensorcnt(0);
#endif
    __syncthreads();

    // scores S = Q * K^T (each wave: 2 16x16 tiles, K=64 -> 2 wmma steps each)
    f32x8 s0 = zero8(), s1 = zero8();
    {
      bf16x16 bk = load_frag(Ks, (wave_n * 2 + 0) * 16, 0, LDP);
      s0 = wmma_bf16(aq0, bk, s0);
      bk = load_frag(Ks, (wave_n * 2 + 0) * 16, 32, LDP);
      s0 = wmma_bf16(aq1, bk, s0);
      bk = load_frag(Ks, (wave_n * 2 + 1) * 16, 0, LDP);
      s1 = wmma_bf16(aq0, bk, s1);
      bk = load_frag(Ks, (wave_n * 2 + 1) * 16, 32, LDP);
      s1 = wmma_bf16(aq1, bk, s1);
    }
#pragma unroll
    for (int r = 0; r < 8; ++r) {  // scale + causal mask -> LDS
      int m = wave_m * 16 + r + hlf * 8;
      int gq = q0 + m;
      int n = (wave_n * 2 + 0) * 16 + nl;
      Sc[m * 65 + n] = (key0 + n <= gq) ? s0[r] * 0.125f : -__builtin_inff();
      n = (wave_n * 2 + 1) * 16 + nl;
      Sc[m * 65 + n] = (key0 + n <= gq) ? s1[r] * 0.125f : -__builtin_inff();
    }
    __syncthreads();

    if (tid < 64) {  // online softmax per query row
      const int m = tid;
      float mold = row_max[m];
      float mt = -__builtin_inff();
#pragma unroll 8
      for (int n = 0; n < 64; ++n) mt = fmaxf(mt, Sc[m * 65 + n]);
      float mnew = fmaxf(mold, mt);
      float sc = __expf(mold - mnew);
      float psum = 0.f;
#pragma unroll 8
      for (int n = 0; n < 64; ++n) {
        float p = __expf(Sc[m * 65 + n] - mnew);
        Ps[m * LDP + n] = f2bf(p);
        psum += p;
      }
      row_sum[m] = row_sum[m] * sc + psum;
      row_max[m] = mnew;
      row_scale[m] = sc;
    }
    __syncthreads();

#pragma unroll
    for (int r = 0; r < 8; ++r) {  // rescale running O
      float sc = row_scale[wave_m * 16 + r + hlf * 8];
      o0[r] *= sc;
      o1[r] *= sc;
    }
    {  // O += P * V
      bf16x16 ap = load_frag(Ps, wave_m * 16, 0, LDP);
      bf16x16 bv = load_frag(Vts, (wave_n * 2 + 0) * 16, 0, LDP);
      o0 = wmma_bf16(ap, bv, o0);
      bv = load_frag(Vts, (wave_n * 2 + 1) * 16, 0, LDP);
      o1 = wmma_bf16(ap, bv, o1);
      ap = load_frag(Ps, wave_m * 16, 32, LDP);
      bv = load_frag(Vts, (wave_n * 2 + 0) * 16, 32, LDP);
      o0 = wmma_bf16(ap, bv, o0);
      bv = load_frag(Vts, (wave_n * 2 + 1) * 16, 32, LDP);
      o1 = wmma_bf16(ap, bv, o1);
    }
    __syncthreads();
  }

#pragma unroll
  for (int r = 0; r < 8; ++r) {
    int m = wave_m * 16 + r + hlf * 8;
    float inv = 1.f / row_sum[m];
    size_t base = (size_t)(q0 + m) * (NHEAD * HD) + h * HD;
    ob[base + (wave_n * 2 + 0) * 16 + nl] = f2bf(o0[r] * inv);
    ob[base + (wave_n * 2 + 1) * 16 + nl] = f2bf(o1[r] * inv);
  }
}

extern "C" void kernel_launch(void* const* d_in, const int* in_sizes, int n_in,
                              void* d_out, int out_size, void* d_ws, size_t ws_size,
                              hipStream_t stream) {
  (void)in_sizes; (void)n_in; (void)out_size; (void)ws_size;
  const float* x     = (const float*)d_in[0];  // [S][H]
  const float* w_qkv = (const float*)d_in[1];  // [3072][H]
  const float* w_o   = (const float*)d_in[2];  // [H][2048]
  float* out = (float*)d_out;                  // [S][H]

  unsigned short* xb    = (unsigned short*)d_ws;
  unsigned short* wqkvb = xb + (size_t)S_LEN * H_DIM;
  unsigned short* wob   = wqkvb + (size_t)QKV_N * H_DIM;
  float* qkv = (float*)(wob + (size_t)H_DIM * H_DIM);
  unsigned short* qbf = (unsigned short*)(qkv + (size_t)S_LEN * QKV_N);
  unsigned short* kbf = qbf + (size_t)S_LEN * NHEAD * HD;
  unsigned short* vbf = kbf + (size_t)KVH * S_LEN * HD;
  unsigned short* abf = vbf + (size_t)KVH * S_LEN * HD;

  {
    int n = S_LEN * H_DIM;
    cvt_f32_bf16_kernel<<<(n + 255) / 256, 256, 0, stream>>>(x, xb, n);
    n = QKV_N * H_DIM;
    cvt_f32_bf16_kernel<<<(n + 255) / 256, 256, 0, stream>>>(w_qkv, wqkvb, n);
    n = H_DIM * H_DIM;
    cvt_f32_bf16_kernel<<<(n + 255) / 256, 256, 0, stream>>>(w_o, wob, n);
  }

  gemm_bf16_kernel<<<dim3(QKV_N / 128, S_LEN / 64), 256, 0, stream>>>(
      xb, wqkvb, qkv, S_LEN, QKV_N, H_DIM);

  {
    int total = S_LEN * 48 * 32;
    rope_split_kernel<<<(total + 255) / 256, 256, 0, stream>>>(qkv, qbf, kbf, vbf);
  }

  flash_attn_kernel<<<dim3(S_LEN / 64, NHEAD), 256, 0, stream>>>(qbf, kbf, vbf, abf);

  gemm_bf16_kernel<<<dim3(H_DIM / 128, S_LEN / 64), 256, 0, stream>>>(
      abf, wob, out, S_LEN, H_DIM, H_DIM);
}
